// MLPblock_adain_81999515615541
// MI455X (gfx1250) — matrix operand order repaired and checked by reading, not compiled, not run on hardware
//
#include <hip/hip_runtime.h>

// ---------------------------------------------------------------------------
// Types for CDNA5 WMMA
// ---------------------------------------------------------------------------
typedef __attribute__((ext_vector_type(16))) __bf16 v16bf;
typedef __attribute__((ext_vector_type(2)))  __bf16 v2bf;
typedef __attribute__((ext_vector_type(8)))  float  v8f;

union FragBF { v16bf v; uint4 q[2]; };

// Native RNE f32->bf16 (lowers to hardware v_cvt_pk_bf16_f32)
__device__ __forceinline__ unsigned short f2bf(float f) {
  __bf16 h = (__bf16)f;
  return __builtin_bit_cast(unsigned short, h);
}
__device__ __forceinline__ unsigned pk2(float a, float b) {
  v2bf p;
  p.x = (__bf16)a;
  p.y = (__bf16)b;
  return __builtin_bit_cast(unsigned, p);
}
__device__ __forceinline__ float silu_f(float x) { return x / (1.0f + __expf(-x)); }
__device__ __forceinline__ float4 silu4(float4 f) {
  f.x = silu_f(f.x); f.y = silu_f(f.y); f.z = silu_f(f.z); f.w = silu_f(f.w);
  return f;
}

// ---------------------------------------------------------------------------
// Generic fused WMMA GEMM:  C[z][m][n] = sum_k A[z][m][k] * BT[z][k][n]  (+epilogue)
//   A : [M,K] row-major, f32 (converted in-register) or bf16 (async DMA)
//   BT: [K,N] row-major bf16 (k-major!) -> staged into LDS [k][n] by pure
//       linear GLOBAL_LOAD_ASYNC_TO_LDS_B128 copies (ASYNCcnt-tracked DMA).
//   Tile: 128x128, BK=32; 256 threads = 8 waves; wave = 32x64 (2x4 WMMA tiles)
//   LDS double-buffered: one barrier per K-step, stage k+1 while WMMA-ing k.
// Epilogue MODE:
//   0: C = acc + biasN[n]
//   1: C = adain(adainIn) + acc + biasN[n]       (stats[4][B][N], b = m/S)
//   2: C = resid + silu(acc + biasM[m])          (seq-mix conv bias over t)
//   3: C = resid + silu(acc + biasN[n])
// ---------------------------------------------------------------------------
template <int ABF, int ASILU, int MODE>
__global__ __launch_bounds__(256, 2)
void wmma_gemm_fused(const void* __restrict__ Aptr, long long a_bs,
                     const unsigned short* __restrict__ BT, long long b_bs,
                     float* __restrict__ C, long long c_bs,
                     int M, int N, int K,
                     const float* __restrict__ biasN, const float* __restrict__ biasM,
                     const float* __restrict__ resid, const float* __restrict__ adainIn,
                     const float* __restrict__ stats, int S)
{
  __shared__ __align__(16) unsigned short As[2][128][40];   // [m][k], 80B rows
  __shared__ __align__(16) unsigned short Bs[2][32][136];   // [k][n], 272B rows

  const int tid  = threadIdx.x;
  const int lane = tid & 31;
  const int wave = tid >> 5;
  const int wm   = wave & 3;    // 4 waves over M (32 rows each)
  const int wn   = wave >> 2;   // 2 waves over N (64 cols each)
  const int lm   = lane & 15;
  const int h    = lane >> 4;

  const int m0 = blockIdx.y * 128;
  const int n0 = blockIdx.x * 128;
  const int z  = blockIdx.z;

  v8f acc[2][4] = {};

  // cooperative tile-load coordinates
  const int ar = tid >> 1;           // A tile row 0..127 (2 threads per 32-k row)
  const int ac = (tid & 1) * 16;     // A k-half
  const int br = tid >> 3;           // B tile k-row 0..31 (8 threads per 128-n row)
  const int bc = (tid & 7) * 16;     // B n-col (16 bf16 = 32B per thread)

  const long long arow = (long long)z * a_bs + (long long)(m0 + ar) * K + ac;
  const long long brow = (long long)z * b_bs + (long long)br * N + n0 + bc;

  uint4 aq0, aq1;   // packed bf16 A staging (f32 path)

  // ---- stage A tile: global -> regs (or async DMA straight into LDS buf) ----
  auto loadA = [&](int k0, int buf) {
    if constexpr (ABF) {
      const unsigned short* Ag = (const unsigned short*)Aptr + arow + k0;
      const unsigned ldst = (unsigned)(size_t)(void*)&As[buf][ar][ac];
      const unsigned long long ga = (unsigned long long)(size_t)Ag;
      asm volatile("global_load_async_to_lds_b128 %0, %1, off"
                   :: "v"(ldst), "v"(ga) : "memory");
      asm volatile("global_load_async_to_lds_b128 %0, %1, off"
                   :: "v"(ldst + 16u), "v"(ga + 16ull) : "memory");
    } else {
      const float* Ag = (const float*)Aptr + arow + k0;
      float4 f0 = *(const float4*)(Ag + 0);
      float4 f1 = *(const float4*)(Ag + 4);
      float4 f2 = *(const float4*)(Ag + 8);
      float4 f3 = *(const float4*)(Ag + 12);
      __builtin_prefetch(Ag + 32, 0, 1);
      if constexpr (ASILU) { f0 = silu4(f0); f1 = silu4(f1); f2 = silu4(f2); f3 = silu4(f3); }
      aq0.x = pk2(f0.x, f0.y); aq0.y = pk2(f0.z, f0.w);
      aq0.z = pk2(f1.x, f1.y); aq0.w = pk2(f1.z, f1.w);
      aq1.x = pk2(f2.x, f2.y); aq1.y = pk2(f2.z, f2.w);
      aq1.z = pk2(f3.x, f3.y); aq1.w = pk2(f3.z, f3.w);
    }
  };
  auto storeA = [&](int buf) {
    if constexpr (!ABF) {
      *(uint4*)&As[buf][ar][ac]     = aq0;
      *(uint4*)&As[buf][ar][ac + 8] = aq1;
    }
  };

  // ---- stage B tile: pure async DMA (BT is k-major bf16, matches LDS layout) ----
  auto loadB = [&](int k0, int buf) {
    const unsigned short* Bg = BT + brow + (long long)k0 * N;
    const unsigned ldst = (unsigned)(size_t)(void*)&Bs[buf][br][bc];
    const unsigned long long ga = (unsigned long long)(size_t)Bg;
    asm volatile("global_load_async_to_lds_b128 %0, %1, off"
                 :: "v"(ldst), "v"(ga) : "memory");
    asm volatile("global_load_async_to_lds_b128 %0, %1, off"
                 :: "v"(ldst + 16u), "v"(ga + 16ull) : "memory");
  };

  // ---- WMMA compute on one LDS buffer ----
  auto compute = [&](int buf) {
    // A layout (ISA 7.12.2): lane holds row M=lane%16; K chunks at h*8 and 16+h*8
    FragBF af[2];
#pragma unroll
    for (int i = 0; i < 2; ++i) {
      const unsigned short* p = &As[buf][wm * 32 + i * 16 + lm][h * 8];
      af[i].q[0] = *(const uint4*)(p);
      af[i].q[1] = *(const uint4*)(p + 16);
    }
#pragma unroll
    for (int j = 0; j < 4; ++j) {
      // B layout: lane holds N=lane%16; K chunks at h*8 and 16+h*8
      FragBF bfr;
      const unsigned short* p = &Bs[buf][h * 8][wn * 64 + j * 16 + lm];
#pragma unroll
      for (int s = 0; s < 16; ++s)
        bfr.v[s] = __builtin_bit_cast(__bf16, p[(s < 8 ? s : s + 8) * 136]);
#pragma unroll
      for (int i = 0; i < 2; ++i) {
        acc[i][j] = __builtin_amdgcn_wmma_f32_16x16x32_bf16(
            false, af[i].v, false, bfr.v, (short)0, acc[i][j], false, false);
      }
    }
  };

  // ---------------- software-pipelined main loop ----------------
  loadA(0, 0); loadB(0, 0);
  storeA(0);
  asm volatile("s_wait_asynccnt 0x0" ::: "memory");
  __syncthreads();

  int st = 0;
  for (int k0 = 32; k0 < K; k0 += 32) {
    loadA(k0, st ^ 1);       // stage next tile (async paths write LDS directly)
    loadB(k0, st ^ 1);
    compute(st);             // WMMAs overlap the in-flight async/global traffic
    storeA(st ^ 1);
    asm volatile("s_wait_asynccnt 0x0" ::: "memory");
    __syncthreads();         // single barrier per K-step
    st ^= 1;
  }
  compute(st);

  // ---------------- fused epilogue ----------------
  const long long czoff = (long long)z * c_bs;
#pragma unroll
  for (int i = 0; i < 2; ++i) {
#pragma unroll
    for (int j = 0; j < 4; ++j) {
      const int tm = m0 + wm * 32 + i * 16 + h * 8;
      const int tn = n0 + wn * 64 + j * 16 + lm;
#pragma unroll
      for (int r = 0; r < 8; ++r) {
        const int gm = tm + r;
        const int gn = tn;
        const long long idx = czoff + (long long)gm * N + gn;
        float v = acc[i][j][r];
        if constexpr (MODE == 0) {
          v += biasN[gn];
        } else if constexpr (MODE == 1) {
          const int BD = (M / S) * N;
          const int bb = gm / S;
          const float cm = stats[0 * BD + bb * N + gn];
          const float cs = stats[1 * BD + bb * N + gn];
          const float sm = stats[2 * BD + bb * N + gn];
          const float ss = stats[3 * BD + bb * N + gn];
          v = (adainIn[idx] - cm) / cs * ss + sm + v + biasN[gn];
        } else if constexpr (MODE == 2) {
          v = resid[idx] + silu_f(v + biasM[gm]);
        } else {
          v = resid[idx] + silu_f(v + biasN[gn]);
        }
        C[idx] = v;
      }
    }
  }
}

// ---------------------------------------------------------------------------
// One-time weight transpose+convert: in f32 [N][K]  ->  out bf16 [K][N]
// ---------------------------------------------------------------------------
__global__ __launch_bounds__(256)
void k_wtrans(const float* __restrict__ in, unsigned short* __restrict__ out,
              int N, int K)
{
  __shared__ float tile[32][33];
  const int k0 = blockIdx.x * 32;
  const int n0 = blockIdx.y * 32;
  const int tx = threadIdx.x, ty = threadIdx.y;   // (32, 8)
#pragma unroll
  for (int i = 0; i < 4; ++i)
    tile[ty + 8 * i][tx] = in[(long long)(n0 + ty + 8 * i) * K + (k0 + tx)];
  __syncthreads();
#pragma unroll
  for (int i = 0; i < 4; ++i)
    out[(long long)(k0 + ty + 8 * i) * N + (n0 + tx)] = f2bf(tile[tx][ty + 8 * i]);
}

// ---------------------------------------------------------------------------
// AdaIN stats: per (b,d), mean & unbiased std over seq axis for content+style
// stats layout: [4][B][D] = c_mean, c_std, s_mean, s_std
// ---------------------------------------------------------------------------
__global__ __launch_bounds__(256)
void k_adain_stats(const float* __restrict__ content, const float* __restrict__ style,
                   float* __restrict__ stats, int B, int S, int D)
{
  const int idx = blockIdx.x * blockDim.x + threadIdx.x;   // 0..B*D-1
  if (idx >= B * D) return;
  const int b = idx / D;
  const int d = idx - b * D;
  const long long base = (long long)b * S * D + d;
  float cs1 = 0.f, cs2 = 0.f, ss1 = 0.f, ss2 = 0.f;
  for (int s = 0; s < S; ++s) {
    const float c = content[base + (long long)s * D];
    const float t = style[base + (long long)s * D];
    cs1 += c; cs2 += c * c;
    ss1 += t; ss2 += t * t;
  }
  const float inv = 1.0f / (float)S;
  const float cmean = cs1 * inv;
  const float smean = ss1 * inv;
  const float cvar = (cs2 - (float)S * cmean * cmean) / (float)(S - 1);
  const float svar = (ss2 - (float)S * smean * smean) / (float)(S - 1);
  const int BD = B * D;
  stats[0 * BD + idx] = cmean;
  stats[1 * BD + idx] = sqrtf(cvar + 1e-5f);
  stats[2 * BD + idx] = smean;
  stats[3 * BD + idx] = sqrtf(svar + 1e-5f);
}

// ---------------------------------------------------------------------------
// Row LayerNorm over D=1024 -> bf16 output  (one block of 256 per row)
// ---------------------------------------------------------------------------
__global__ __launch_bounds__(256)
void k_layernorm(const float* __restrict__ x, const float* __restrict__ g,
                 const float* __restrict__ bta, unsigned short* __restrict__ out, int D)
{
  __shared__ float red[256];
  const long long row = blockIdx.x;
  const int t = threadIdx.x;
  const float* xp = x + row * D;
  float lx[4];
  float s = 0.f;
#pragma unroll
  for (int i = 0; i < 4; ++i) { lx[i] = xp[t + 256 * i]; s += lx[i]; }
  red[t] = s; __syncthreads();
  for (int o = 128; o > 0; o >>= 1) { if (t < o) red[t] += red[t + o]; __syncthreads(); }
  const float mu = red[0] / (float)D;
  __syncthreads();
  float v = 0.f;
#pragma unroll
  for (int i = 0; i < 4; ++i) { const float dlt = lx[i] - mu; v += dlt * dlt; }
  red[t] = v; __syncthreads();
  for (int o = 128; o > 0; o >>= 1) { if (t < o) red[t] += red[t + o]; __syncthreads(); }
  const float rstd = rsqrtf(red[0] / (float)D + 1e-5f);
#pragma unroll
  for (int i = 0; i < 4; ++i) {
    const int c = t + 256 * i;
    out[row * D + c] = f2bf((lx[i] - mu) * rstd * g[c] + bta[c]);
  }
}

// ---------------------------------------------------------------------------
// float4 copy
// ---------------------------------------------------------------------------
__global__ __launch_bounds__(256)
void k_copyf(const float* __restrict__ src, float* __restrict__ dst, long long n4)
{
  const float4* s4 = (const float4*)src;
  float4* d4 = (float4*)dst;
  for (long long i = blockIdx.x * (long long)blockDim.x + threadIdx.x; i < n4;
       i += (long long)gridDim.x * blockDim.x)
    d4[i] = s4[i];
}

// ---------------------------------------------------------------------------
extern "C" void kernel_launch(void* const* d_in, const int* in_sizes, int n_in,
                              void* d_out, int out_size, void* d_ws, size_t ws_size,
                              hipStream_t stream) {
  const float* x     = (const float*)d_in[0];
  const float* embed = (const float*)d_in[1];
  const float* style = (const float*)d_in[2];
  const float* Wc    = (const float*)d_in[3];
  const float* bc    = (const float*)d_in[4];
  const float* We    = (const float*)d_in[5];
  const float* be    = (const float*)d_in[6];
  const float* W0    = (const float*)d_in[7];
  const float* b0    = (const float*)d_in[8];
  const float* W1    = (const float*)d_in[9];
  const float* b1    = (const float*)d_in[10];
  const float* g0    = (const float*)d_in[11];
  const float* beta0 = (const float*)d_in[12];
  const float* g1    = (const float*)d_in[13];
  const float* beta1 = (const float*)d_in[14];
  float* out = (float*)d_out;

  const int B = 64, S = 256, D = 1024;
  const long long BS  = (long long)B * S;    // 16384
  const long long BSD = BS * D;              // 16,777,216

  // workspace layout (~169 MB)
  char* ws = (char*)d_ws;
  float* adain_in      = (float*)ws;                              // BSD f32
  float* xcur          = adain_in + BSD;                          // BSD f32
  float* stats         = xcur + BSD;                              // 4*B*D f32
  unsigned short* lnb  = (unsigned short*)(stats + 4LL * B * D);  // BSD bf16
  unsigned short* WcT  = lnb + BSD;                               // [2D][D] bf16
  unsigned short* WeT  = WcT + 2LL * D * D;                       // [D][D] bf16
  unsigned short* W1T  = WeT + (long long)D * D;                  // [D][D] bf16

  dim3 blk(256);
  dim3 tblk(32, 8);

  // 0) one-time: weights f32 [N][K] -> bf16 [K][N] (k-major, async-DMA friendly)
  k_wtrans<<<dim3(2 * D / 32, D / 32), tblk, 0, stream>>>(Wc, WcT, D, 2 * D);
  k_wtrans<<<dim3(D / 32, D / 32), tblk, 0, stream>>>(We, WeT, D, D);
  k_wtrans<<<dim3(D / 32, D / 32), tblk, 0, stream>>>(W1, W1T, D, D);

  // 1) adain_in = x @ Wc^T + bc          [16384,2048]x[1024,2048]^T
  wmma_gemm_fused<0, 0, 0><<<dim3(D / 128, (int)(BS / 128), 1), blk, 0, stream>>>(
      x, 0LL, WcT, 0LL, adain_in, 0LL,
      (int)BS, D, 2 * D, bc, nullptr, nullptr, nullptr, nullptr, S);

  // 2) AdaIN statistics over seq axis
  k_adain_stats<<<(B * D) / 256, blk, 0, stream>>>(adain_in, style, stats, B, S, D);

  // 3) xcur = adain(adain_in) + silu(embed) @ We^T + be
  wmma_gemm_fused<0, 1, 1><<<dim3(D / 128, (int)(BS / 128), 1), blk, 0, stream>>>(
      embed, 0LL, WeT, 0LL, xcur, 0LL,
      (int)BS, D, D, be, nullptr, nullptr, adain_in, stats, S);

  // 4) LN0 -> bf16  (ln[b][s][d] is already k-major for the seq-mix GEMM)
  k_layernorm<<<(int)BS, blk, 0, stream>>>(xcur, g0, beta0, lnb, D);

  // 5) xcur += silu(W0 x ln0 + b0[t])    batched over B, M=K=S=256, N=D
  wmma_gemm_fused<0, 0, 2><<<dim3(D / 128, S / 128, B), blk, 0, stream>>>(
      W0, 0LL, lnb, (long long)S * D, xcur, (long long)S * D,
      S, D, S, nullptr, b0, xcur, nullptr, nullptr, S);

  // 6) LN1 -> bf16
  k_layernorm<<<(int)BS, blk, 0, stream>>>(xcur, g1, beta1, lnb, D);

  // 7) out[0..BSD) = xcur + silu(ln1 @ W1^T + b1)   (A also async-to-LDS DMA)
  wmma_gemm_fused<1, 0, 3><<<dim3(D / 128, (int)(BS / 128), 1), blk, 0, stream>>>(
      lnb, 0LL, W1T, 0LL, out, 0LL,
      (int)BS, D, D, b1, nullptr, xcur, nullptr, nullptr, S);

  // 8) passthrough outputs: embed, style_code
  k_copyf<<<2048, blk, 0, stream>>>(embed, out + BSD, BSD / 4);
  k_copyf<<<2048, blk, 0, stream>>>(style, out + 2 * BSD, BSD / 4);
}
